// EncoderLayer_35811437314054
// MI455X (gfx1250) — matrix operand order repaired and checked by reading, not compiled, not run on hardware
//
#include <hip/hip_runtime.h>
#include <hip/hip_bf16.h>
#include <math.h>

// ---------------------------------------------------------------------------
// Transformer encoder layer for MI455X (gfx1250), bf16 WMMA + f32 accumulate.
// ---------------------------------------------------------------------------

typedef __attribute__((ext_vector_type(16))) __bf16 v16bf;
typedef __attribute__((ext_vector_type(8)))  __bf16 v8bf;
typedef __attribute__((ext_vector_type(8)))  float  v8f;

#define D_MODEL 1024
#define N_HEADS 16
#define D_K     64
#define D_FF    4096
#define SEQ     2048
#define BATCH   2
#define NTOK    (BATCH * SEQ)

static __device__ __forceinline__ v8f wmma_bf16(v16bf a, v16bf b, v8f c) {
  // D = A(16x32 bf16) * B(32x16 bf16) + C(16x16 f32)
  return __builtin_amdgcn_wmma_f32_16x16x32_bf16(
      /*neg_a=*/false, a, /*neg_b=*/false, b,
      /*c_mod=*/(short)0, c, /*reuse_a=*/false, /*reuse_b=*/false);
}

static __device__ __forceinline__ v16bf cat8(v8bf lo, v8bf hi) {
  return __builtin_shufflevector(lo, hi, 0, 1, 2, 3, 4, 5, 6, 7,
                                 8, 9, 10, 11, 12, 13, 14, 15);
}

// ---------------------------------------------------------------------------
// Weight transpose + fp32 -> bf16 convert:  src[in][out] -> dst[out][in]
// ---------------------------------------------------------------------------
__global__ void wt_kernel(const float* __restrict__ src, __bf16* __restrict__ dst,
                          int in_dim, int out_dim) {
  size_t total = (size_t)in_dim * out_dim;
  size_t stride = (size_t)gridDim.x * blockDim.x;
  for (size_t idx = (size_t)blockIdx.x * blockDim.x + threadIdx.x; idx < total;
       idx += stride) {
    int i = (int)(idx / (size_t)out_dim);
    int o = (int)(idx % (size_t)out_dim);
    dst[(size_t)o * in_dim + i] = (__bf16)src[idx];
  }
}

// ---------------------------------------------------------------------------
// LayerNorm with Bessel-corrected std, out = w*(x-mean)/(std+eps)+b  -> bf16
// One 256-thread block per row of 1024.
// ---------------------------------------------------------------------------
__global__ void ln_kernel(const float* __restrict__ x, const float* __restrict__ w,
                          const float* __restrict__ bias, __bf16* __restrict__ out) {
  __shared__ float red[256];
  const int t = threadIdx.x;
  const size_t row = blockIdx.x;
  const float* xr = x + row * D_MODEL;
  float4 xv = reinterpret_cast<const float4*>(xr)[t];

  red[t] = xv.x + xv.y + xv.z + xv.w;
  __syncthreads();
  for (int o = 128; o > 0; o >>= 1) {
    if (t < o) red[t] += red[t + o];
    __syncthreads();
  }
  const float mean = red[0] * (1.0f / D_MODEL);
  __syncthreads();

  const float d0 = xv.x - mean, d1 = xv.y - mean, d2 = xv.z - mean, d3 = xv.w - mean;
  red[t] = d0 * d0 + d1 * d1 + d2 * d2 + d3 * d3;
  __syncthreads();
  for (int o = 128; o > 0; o >>= 1) {
    if (t < o) red[t] += red[t + o];
    __syncthreads();
  }
  const float var = red[0] * (1.0f / (D_MODEL - 1));
  const float inv = 1.0f / (sqrtf(var) + 1e-6f);

  float4 wv = reinterpret_cast<const float4*>(w)[t];
  float4 bv = reinterpret_cast<const float4*>(bias)[t];
  __bf16* orow = out + row * D_MODEL + (size_t)t * 4;
  orow[0] = (__bf16)(d0 * inv * wv.x + bv.x);
  orow[1] = (__bf16)(d1 * inv * wv.y + bv.y);
  orow[2] = (__bf16)(d2 * inv * wv.z + bv.z);
  orow[3] = (__bf16)(d3 * inv * wv.w + bv.w);
}

// ---------------------------------------------------------------------------
// Generic bf16 WMMA GEMM:  C[M,N] = A[M,K] @ W[N,K]^T + bias
// One wave (32 threads) computes a 16x64 output strip (A reused over 4 tiles).
// Optional exact-erf GELU, optional fp32 residual add; fp32 and/or bf16 out.
// ---------------------------------------------------------------------------
__global__ void gemm_kernel(const __bf16* __restrict__ A, const __bf16* __restrict__ W,
                            const float* __restrict__ bias,
                            const float* __restrict__ resid,
                            float* __restrict__ out_f, __bf16* __restrict__ out_b,
                            int M, int N, int K, int act) {
  const int lane = threadIdx.x;
  const int col  = lane & 15;
  const int hi   = lane >> 4;     // 0 or 1 (lane half)
  const int hi8  = hi * 8;
  const int kb16 = hi * 16;
  const int m0 = blockIdx.x * 16;
  const int n0 = blockIdx.y * 64;

  const __bf16* arow = A + (size_t)(m0 + col) * K;
  const __bf16* wrow = W + (size_t)(n0 + col) * K;

  v8f acc[4] = {};
  for (int kk = 0; kk < K; kk += 32) {
    // A layout: lane holds row m=col; elems = k(kb16..kb16+15) contiguous
    v16bf a = *reinterpret_cast<const v16bf*>(arow + kk + kb16);
#pragma unroll
    for (int s = 0; s < 4; ++s) {
      // B layout: lane holds col n; elems = k(kb16..kb16+15) contiguous in W^T
      v16bf bm = *reinterpret_cast<const v16bf*>(wrow + (size_t)s * 16 * K + kk + kb16);
      acc[s] = wmma_bf16(a, bm, acc[s]);
    }
  }

#pragma unroll
  for (int s = 0; s < 4; ++s) {
    const int n = n0 + s * 16 + col;
    const float bs = bias[n];
#pragma unroll
    for (int r = 0; r < 8; ++r) {
      const size_t m = (size_t)(m0 + r + hi8);
      float vv = acc[s][r] + bs;
      if (act) vv = 0.5f * vv * (1.0f + erff(vv * 0.70710678118654752f));
      if (resid) vv += resid[m * N + n];
      if (out_f) out_f[m * N + n] = vv;
      if (out_b) out_b[m * N + n] = (__bf16)vv;
    }
  }
}

// ---------------------------------------------------------------------------
// Flash attention. One wave per (batch*head, 16-query tile).
// Scores computed transposed: St = K_tile(A) x Q^T(B)  ->  St[key][query],
// so exp(St) is already in the A-operand layout for the P @ V WMMA.
// q,k,v,out are bf16 [token][h*64+d].
// ---------------------------------------------------------------------------
__global__ void attn_kernel(const __bf16* __restrict__ q, const __bf16* __restrict__ k,
                            const __bf16* __restrict__ v, const int* __restrict__ mask,
                            __bf16* __restrict__ out) {
  const int lane = threadIdx.x;
  const int col  = lane & 15;
  const int hi   = lane >> 4;
  const int hi8  = hi * 8;
  const int kb16 = hi * 16;
  const int qt = blockIdx.x;             // query tile, 16 rows
  const int b  = blockIdx.y >> 4;        // N_HEADS == 16
  const int h  = blockIdx.y & 15;
  const int tokbase = b * SEQ;
  const int qbase = tokbase + qt * 16;
  const size_t hd = (size_t)h * D_K;

  // Q as the B operand (col = query), both d_k halves, loaded once.
  const __bf16* qrow = q + (size_t)(qbase + col) * D_MODEL + hd;
  const v16bf qb0 = cat8(*reinterpret_cast<const v8bf*>(qrow + hi8),
                         *reinterpret_cast<const v8bf*>(qrow + 16 + hi8));
  const v16bf qb1 = cat8(*reinterpret_cast<const v8bf*>(qrow + 32 + hi8),
                         *reinterpret_cast<const v8bf*>(qrow + 48 + hi8));

  v8f acc[4] = {};                       // O tile 16x64 (4 d-slices of 16)
  float m_i = -INFINITY, l_i = 0.0f;
  const float scale = 0.125f;            // 1/sqrt(64)
  const int* mrow = mask + b * SEQ;

  for (int c = 0; c < SEQ; c += 32) {
    const __bf16* kr0 = k + (size_t)(tokbase + c + col) * D_MODEL + hd;
    const __bf16* kr1 = kr0 + (size_t)16 * D_MODEL;

    v8f st0 = {}, st1 = {};
    v16bf ka;
    ka = cat8(*reinterpret_cast<const v8bf*>(kr0 + hi8),
              *reinterpret_cast<const v8bf*>(kr0 + 16 + hi8));
    st0 = wmma_bf16(ka, qb0, st0);
    ka = cat8(*reinterpret_cast<const v8bf*>(kr0 + 32 + hi8),
              *reinterpret_cast<const v8bf*>(kr0 + 48 + hi8));
    st0 = wmma_bf16(ka, qb1, st0);
    ka = cat8(*reinterpret_cast<const v8bf*>(kr1 + hi8),
              *reinterpret_cast<const v8bf*>(kr1 + 16 + hi8));
    st1 = wmma_bf16(ka, qb0, st1);
    ka = cat8(*reinterpret_cast<const v8bf*>(kr1 + 32 + hi8),
              *reinterpret_cast<const v8bf*>(kr1 + 48 + hi8));
    st1 = wmma_bf16(ka, qb1, st1);

    // St layout: lane = query col; element r = key (c + r + hi8) [+16 for st1]
    float s0[8], s1[8];
    float mc = -INFINITY;
#pragma unroll
    for (int r = 0; r < 8; ++r) {
      const int key = c + r + hi8;
      s0[r] = (mrow[key] != 0)      ? st0[r] * scale : -INFINITY;
      s1[r] = (mrow[key + 16] != 0) ? st1[r] * scale : -INFINITY;
      mc = fmaxf(mc, fmaxf(s0[r], s1[r]));
    }
    mc = fmaxf(mc, __shfl_xor(mc, 16, 32));
    const float m_new = fmaxf(m_i, mc);
    const float corr = __expf(m_i - m_new);

    float psum = 0.0f;
    v16bf pa;                              // P in A-operand layout, no shuffles
#pragma unroll
    for (int r = 0; r < 8; ++r) {
      const float e0 = __expf(s0[r] - m_new);
      const float e1 = __expf(s1[r] - m_new);
      psum += e0 + e1;
      pa[r]     = (__bf16)e0;
      pa[r + 8] = (__bf16)e1;
    }
    psum += __shfl_xor(psum, 16, 32);
    l_i = l_i * corr + psum;
    m_i = m_new;

    float rowcorr[8];
#pragma unroll
    for (int r = 0; r < 8; ++r) rowcorr[r] = __shfl(corr, r + hi8, 32);

    const __bf16* vbase = v + (size_t)(tokbase + c + hi8) * D_MODEL + hd + col;
#pragma unroll
    for (int s = 0; s < 4; ++s) {
      v16bf vb;   // B operand: lane = d col (slice*16+col), elems = 16 keys
#pragma unroll
      for (int e = 0; e < 16; ++e)
        vb[e] = vbase[(size_t)(e + (e >> 3) * 8) * D_MODEL + s * 16];
      v8f a = acc[s];
#pragma unroll
      for (int r = 0; r < 8; ++r) a[r] *= rowcorr[r];
      acc[s] = wmma_bf16(pa, vb, a);
    }
  }

  const float inv = 1.0f / l_i;
  float rowinv[8];
#pragma unroll
  for (int r = 0; r < 8; ++r) rowinv[r] = __shfl(inv, r + hi8, 32);
#pragma unroll
  for (int s = 0; s < 4; ++s)
#pragma unroll
    for (int r = 0; r < 8; ++r)
      out[(size_t)(qbase + r + hi8) * D_MODEL + hd + s * 16 + col] =
          (__bf16)(acc[s][r] * rowinv[r]);
}

// ---------------------------------------------------------------------------
// Host orchestration
// ---------------------------------------------------------------------------
extern "C" void kernel_launch(void* const* d_in, const int* in_sizes, int n_in,
                              void* d_out, int out_size, void* d_ws, size_t ws_size,
                              hipStream_t stream) {
  (void)in_sizes; (void)n_in; (void)out_size; (void)ws_size;
  const float* x    = (const float*)d_in[0];
  const int*   mask = (const int*)d_in[1];
  const float* wq = (const float*)d_in[2];   const float* bq = (const float*)d_in[3];
  const float* wk = (const float*)d_in[4];   const float* bk = (const float*)d_in[5];
  const float* wv = (const float*)d_in[6];   const float* bv = (const float*)d_in[7];
  const float* wo = (const float*)d_in[8];   const float* bo = (const float*)d_in[9];
  const float* ln1w = (const float*)d_in[10]; const float* ln1b = (const float*)d_in[11];
  const float* ln2w = (const float*)d_in[12]; const float* ln2b = (const float*)d_in[13];
  const float* w1 = (const float*)d_in[14];  const float* b1 = (const float*)d_in[15];
  const float* w2 = (const float*)d_in[16];  const float* b2 = (const float*)d_in[17];
  float* out = (float*)d_out;

  // Workspace layout (80 MiB peak, ffn1 overlays the q/k/v/attn region).
  char* ws = (char*)d_ws;
  const size_t MiB = (size_t)1 << 20;
  __bf16* wq_t = (__bf16*)(ws + 0 * MiB);   // [1024][1024]
  __bf16* wk_t = (__bf16*)(ws + 2 * MiB);
  __bf16* wv_t = (__bf16*)(ws + 4 * MiB);
  __bf16* wo_t = (__bf16*)(ws + 6 * MiB);
  __bf16* w1_t = (__bf16*)(ws + 8 * MiB);   // [4096][1024]
  __bf16* w2_t = (__bf16*)(ws + 16 * MiB);  // [1024][4096]
  __bf16* xln  = (__bf16*)(ws + 24 * MiB);  // [4096][1024]; reused as g after attn
  __bf16* qbuf = (__bf16*)(ws + 32 * MiB);  // [4096][1024]
  __bf16* kbuf = (__bf16*)(ws + 40 * MiB);
  __bf16* vbuf = (__bf16*)(ws + 48 * MiB);
  __bf16* abuf = (__bf16*)(ws + 56 * MiB);  // attention output
  __bf16* f1   = (__bf16*)(ws + 32 * MiB);  // ffn1 [4096][4096], overlays qkv+attn
  float*  hbuf = (float*)(ws + 64 * MiB);   // residual h [4096][1024] fp32

  // 1. weight transpose + convert (bf16 [N][K])
  wt_kernel<<<4096, 256, 0, stream>>>(wq, wq_t, D_MODEL, D_MODEL);
  wt_kernel<<<4096, 256, 0, stream>>>(wk, wk_t, D_MODEL, D_MODEL);
  wt_kernel<<<4096, 256, 0, stream>>>(wv, wv_t, D_MODEL, D_MODEL);
  wt_kernel<<<4096, 256, 0, stream>>>(wo, wo_t, D_MODEL, D_MODEL);
  wt_kernel<<<4096, 256, 0, stream>>>(w1, w1_t, D_MODEL, D_FF);
  wt_kernel<<<4096, 256, 0, stream>>>(w2, w2_t, D_FF, D_MODEL);

  // 2. LN1
  ln_kernel<<<NTOK, 256, 0, stream>>>(x, ln1w, ln1b, xln);

  // 3. QKV projections
  dim3 g_mm(NTOK / 16, D_MODEL / 64);
  gemm_kernel<<<g_mm, 32, 0, stream>>>(xln, wq_t, bq, nullptr, nullptr, qbuf,
                                       NTOK, D_MODEL, D_MODEL, 0);
  gemm_kernel<<<g_mm, 32, 0, stream>>>(xln, wk_t, bk, nullptr, nullptr, kbuf,
                                       NTOK, D_MODEL, D_MODEL, 0);
  gemm_kernel<<<g_mm, 32, 0, stream>>>(xln, wv_t, bv, nullptr, nullptr, vbuf,
                                       NTOK, D_MODEL, D_MODEL, 0);

  // 4. flash attention
  attn_kernel<<<dim3(SEQ / 16, BATCH * N_HEADS), 32, 0, stream>>>(
      qbuf, kbuf, vbuf, mask, abuf);

  // 5. output projection + residual (h = x + attn @ Wo + bo), fp32
  gemm_kernel<<<g_mm, 32, 0, stream>>>(abuf, wo_t, bo, x, hbuf, nullptr,
                                       NTOK, D_MODEL, D_MODEL, 0);

  // 6. LN2 (reuse xln buffer for g)
  ln_kernel<<<NTOK, 256, 0, stream>>>(hbuf, ln2w, ln2b, xln);

  // 7. FFN1 with exact GELU
  gemm_kernel<<<dim3(NTOK / 16, D_FF / 64), 32, 0, stream>>>(
      xln, w1_t, b1, nullptr, nullptr, f1, NTOK, D_FF, D_MODEL, 1);

  // 8. FFN2 + residual -> d_out
  gemm_kernel<<<g_mm, 32, 0, stream>>>(f1, w2_t, b2, hbuf, out, nullptr,
                                       NTOK, D_MODEL, D_FF, 0);
}
